// Mamba_Prior_1632087572668
// MI455X (gfx1250) — matrix operand order repaired and checked by reading, not compiled, not run on hardware
//
#include <hip/hip_runtime.h>
#include <math.h>

typedef __attribute__((ext_vector_type(16))) _Float16 v16h;
typedef __attribute__((ext_vector_type(8)))  float    v8f;

#define LN_EPS 1e-5f
#define BN_EPS 1e-5f

// ---------------------------------------------------------------------------
// Common epilogue: act(v + bias) (+ resid)
// ---------------------------------------------------------------------------
__device__ __forceinline__ float gemm_epilogue(float v, const float* bias,
                                               const float* resid, size_t mn,
                                               int n, int act) {
  if (bias) v += bias[n];
  if (act == 1) {
    v = 0.5f * v * (1.f + erff(v * 0.70710678118f));          // gelu (exact)
  } else if (act == 2) {
    float sp = __logf(1.f + __expf(v));                       // softplus, branch-free
    v = (v > 20.f) ? v : sp;
  }
  if (resid) v += resid[mn];
  return v;
}

// ---------------------------------------------------------------------------
// FULL-tile WMMA GEMM (M%128==0, N%32==0, K%32==0, lda==K).
// 128x32 block tile, 8 waves; each wave: one 16-row stripe x two 16x16 N-tiles
// (2 WMMAs/step, A-fragment reused). Double-buffered LDS fed by CDNA5
// global_load_async_to_lds_b128 (ASYNCcnt): tile k+1 streams in under tile k's
// WMMAs; s_wait_asynccnt 5 drains only the older tile's 5 copies.
// ---------------------------------------------------------------------------
#define APAD 36   // floats per LDS row (32 data + 4 pad): 144B stride, 16B aligned
__global__ __launch_bounds__(256)
void k_gemm_wmma_full(const float* __restrict__ A, const float* __restrict__ Wt,
                      const float* __restrict__ bias, const float* __restrict__ resid,
                      float* __restrict__ out, int M, int N, int K, int act) {
  __shared__ __align__(16) float Asf[2][128][APAD];
  __shared__ __align__(16) float Bsf[2][32][APAD];
  const int tid   = threadIdx.x;
  const int m0    = blockIdx.x * 128;
  const int n0    = blockIdx.y * 32;
  const int wave  = tid >> 5, lane = tid & 31;
  const int half_ = lane >> 4, id16 = lane & 15;

  // copy assignment: 16B chunks; A = 4 rounds (128 rows), B = 1 round (32 rows)
  const int crow = tid >> 3;          // 0..31
  const int ccol = (tid & 7) * 4;     // 0,4,...,28

  auto issue = [&](int buf, int k0) {
#pragma unroll
    for (int r = 0; r < 4; ++r) {
      int row = crow + r * 32;
      const float* ga = A + (size_t)(m0 + row) * K + k0 + ccol;
      uint32_t lds = (uint32_t)(uintptr_t)&Asf[buf][row][ccol];
      asm volatile("global_load_async_to_lds_b128 %0, %1, off"
                   :: "v"(lds), "v"(ga) : "memory");
    }
    const float* gb = Wt + (size_t)(n0 + crow) * K + k0 + ccol;
    uint32_t ldsb = (uint32_t)(uintptr_t)&Bsf[buf][crow][ccol];
    asm volatile("global_load_async_to_lds_b128 %0, %1, off"
                 :: "v"(ldsb), "v"(gb) : "memory");
  };

  v8f acc0 = {}, acc1 = {};
  const int nk = K >> 5;
  issue(0, 0);
  for (int kt = 0; kt < nk; ++kt) {
    const int cur = kt & 1;
    if (kt + 1 < nk) {
      issue(cur ^ 1, (kt + 1) << 5);
      asm volatile("s_wait_asynccnt 5" ::: "memory");   // drain older tile only
    } else {
      asm volatile("s_wait_asynccnt 0" ::: "memory");
    }
    __syncthreads();

    v16h afrag, bfrag0, bfrag1;
    const int arow = wave * 16 + id16;
#pragma unroll
    for (int j = 0; j < 8; ++j) {
      int ka = ((j & 4) ? 16 : 0) + half_ * 8 + 2 * (j & 3);
      afrag[2 * j]      = (_Float16)Asf[cur][arow][ka];
      afrag[2 * j + 1]  = (_Float16)Asf[cur][arow][ka + 1];
      int kb = ((j & 4) ? 16 : 0) + half_ * 8 + 2 * (j & 3);
      bfrag0[2 * j]     = (_Float16)Bsf[cur][id16][kb];
      bfrag0[2 * j + 1] = (_Float16)Bsf[cur][id16][kb + 1];
      bfrag1[2 * j]     = (_Float16)Bsf[cur][16 + id16][kb];
      bfrag1[2 * j + 1] = (_Float16)Bsf[cur][16 + id16][kb + 1];
    }
    acc0 = __builtin_amdgcn_wmma_f32_16x16x32_f16(false, afrag, false, bfrag0,
                                                  (short)0, acc0, false, false);
    acc1 = __builtin_amdgcn_wmma_f32_16x16x32_f16(false, afrag, false, bfrag1,
                                                  (short)0, acc1, false, false);
    __syncthreads();
  }

  const int na = n0 + id16;
  const int nb = n0 + 16 + id16;
#pragma unroll
  for (int r = 0; r < 8; ++r) {
    int m = m0 + wave * 16 + (half_ ? 8 : 0) + r;
    size_t mna = (size_t)m * N + na;
    size_t mnb = (size_t)m * N + nb;
    out[mna] = gemm_epilogue(acc0[r], bias, resid, mna, na, act);
    out[mnb] = gemm_epilogue(acc1[r], bias, resid, mnb, nb, act);
  }
}

// ---------------------------------------------------------------------------
// EDGE WMMA GEMM: arbitrary M/N/K/lda; branchless clamped loads, f16 LDS.
// ---------------------------------------------------------------------------
__global__ __launch_bounds__(256)
void k_gemm_wmma_edge(const float* __restrict__ A, const float* __restrict__ Wt,
                      const float* __restrict__ bias, const float* __restrict__ resid,
                      float* __restrict__ out, int M, int N, int K, int lda, int act) {
  __shared__ _Float16 As[64][40];
  __shared__ _Float16 Bs[32][40];
  const int tid   = threadIdx.x;
  const int m0    = blockIdx.x * 64;
  const int n0    = blockIdx.y * 32;
  const int wave  = tid >> 5, lane = tid & 31;
  const int wm    = wave >> 1, wn = wave & 1;
  const int half_ = lane >> 4, id16 = lane & 15;

  v8f acc = {};
  for (int k0 = 0; k0 < K; k0 += 32) {
    float va[8];
#pragma unroll
    for (int e = 0; e < 8; ++e) {
      int idx = e * 256 + tid;
      int row = idx >> 5, kk = idx & 31;
      int m = m0 + row, k = k0 + kk;
      bool ok = (m < M) & (k < K);
      float t = A[ok ? ((size_t)m * lda + k) : 0];
      va[e] = ok ? t : 0.f;
    }
    float vb[4];
#pragma unroll
    for (int e = 0; e < 4; ++e) {
      int idx = e * 256 + tid;
      int nr = idx >> 5, kk = idx & 31;
      int n = n0 + nr, k = k0 + kk;
      bool ok = (n < N) & (k < K);
      float t = Wt[ok ? ((size_t)n * K + k) : 0];
      vb[e] = ok ? t : 0.f;
    }
#pragma unroll
    for (int e = 0; e < 8; ++e) {
      int idx = e * 256 + tid;
      As[idx >> 5][idx & 31] = (_Float16)va[e];
    }
#pragma unroll
    for (int e = 0; e < 4; ++e) {
      int idx = e * 256 + tid;
      Bs[idx >> 5][idx & 31] = (_Float16)vb[e];
    }
    if (k0 + 32 < K)
      __builtin_prefetch(A + (size_t)(m0 + (tid >> 5)) * lda + k0 + 32, 0, 1);
    __syncthreads();

    v16h afrag, bfrag;
    const int arow = wm * 16 + id16;
    const int bcol = wn * 16 + id16;
#pragma unroll
    for (int j = 0; j < 8; ++j) {
      int ka = ((j & 4) ? 16 : 0) + half_ * 8 + 2 * (j & 3);
      afrag[2 * j]     = As[arow][ka];
      afrag[2 * j + 1] = As[arow][ka + 1];
      int kb = ((j & 4) ? 16 : 0) + half_ * 8 + 2 * (j & 3);
      bfrag[2 * j]     = Bs[bcol][kb];
      bfrag[2 * j + 1] = Bs[bcol][kb + 1];
    }
    acc = __builtin_amdgcn_wmma_f32_16x16x32_f16(false, afrag, false, bfrag,
                                                 (short)0, acc, false, false);
    __syncthreads();
  }

  const int n = n0 + wn * 16 + id16;
#pragma unroll
  for (int r = 0; r < 8; ++r) {
    int m = m0 + wm * 16 + (half_ ? 8 : 0) + r;
    if (m < M && n < N) {
      size_t mn = (size_t)m * N + n;
      out[mn] = gemm_epilogue(acc[r], bias, resid, mn, n, act);
    }
  }
}

// ---------------------------------------------------------------------------
// LayerNorm over last dim (one wave per token), optional SiLU gate multiply.
// ---------------------------------------------------------------------------
__global__ void k_layernorm(const float* __restrict__ x, const float* __restrict__ g,
                            const float* __restrict__ b, const float* __restrict__ gate,
                            int gate_ld, int gate_off, float* __restrict__ y,
                            int T, int C) {
  int wave = (blockIdx.x * blockDim.x + threadIdx.x) >> 5;
  int lane = threadIdx.x & 31;
  if (wave >= T) return;
  const float* xr = x + (size_t)wave * C;
  float s = 0.f, ss = 0.f;
  for (int c = lane; c < C; c += 32) { float v = xr[c]; s += v; ss += v * v; }
  for (int o = 16; o; o >>= 1) { s += __shfl_xor(s, o); ss += __shfl_xor(ss, o); }
  float mean = s / C;
  float var  = ss / C - mean * mean;
  float rstd = rsqrtf(var + LN_EPS);
  float* yr = y + (size_t)wave * C;
  const float* gr = gate ? gate + (size_t)wave * gate_ld + gate_off : nullptr;
  for (int c = lane; c < C; c += 32) {
    float v = (xr[c] - mean) * rstd * g[c] + b[c];
    if (gr) { float z = gr[c]; v *= z / (1.f + __expf(-z)); }
    yr[c] = v;
  }
}

// ---------------------------------------------------------------------------
// Depthwise 3x3 (pad 1) + SiLU, NHWC; input is x-half of xz (stride 2D).
// ---------------------------------------------------------------------------
__global__ void k_dwconv_silu(const float* __restrict__ xz, const float* __restrict__ w,
                              const float* __restrict__ bias, float* __restrict__ out,
                              int Bn, int H, int W, int D) {
  int i = blockIdx.x * blockDim.x + threadIdx.x;
  int total = Bn * H * W * D;
  if (i >= total) return;
  int d  = i % D;
  int w_ = (i / D) % W;
  int h  = (i / (D * W)) % H;
  int b  = i / (D * W * H);
  int twoD = 2 * D;
  float acc = bias[d];
#pragma unroll
  for (int ii = 0; ii < 3; ++ii) {
#pragma unroll
    for (int jj = 0; jj < 3; ++jj) {
      int hi = h + ii - 1, wi = w_ + jj - 1;
      if (hi >= 0 && hi < H && wi >= 0 && wi < W)
        acc += xz[((size_t)((b * H + hi) * W + wi)) * twoD + d] * w[d * 9 + ii * 3 + jj];
    }
  }
  out[i] = acc / (1.f + __expf(-acc));   // silu
}

// ---------------------------------------------------------------------------
// Pack conv output into 4 direction-ordered copies: xsT[(k*Bn+b)*L + t][d]
// ---------------------------------------------------------------------------
__global__ void k_pack4(const float* __restrict__ xc, float* __restrict__ xsT,
                        int Bn, int H, int W, int D) {
  int i = blockIdx.x * blockDim.x + threadIdx.x;
  int L = H * W;
  int total = Bn * L * D;
  if (i >= total) return;
  int d = i % D;
  int l = (i / D) % L;
  int b = i / (D * L);
  int h = l / W, w_ = l % W;
  int t1 = w_ * H + h;
  float v = xc[i];
  xsT[((size_t)(0 * Bn + b) * L + l)            * D + d] = v;
  xsT[((size_t)(1 * Bn + b) * L + t1)           * D + d] = v;
  xsT[((size_t)(2 * Bn + b) * L + (L - 1 - l))  * D + d] = v;
  xsT[((size_t)(3 * Bn + b) * L + (L - 1 - t1)) * D + d] = v;
}

// ---------------------------------------------------------------------------
// Selective scan: one block per (k,b); one thread per channel d; 16-wide state
// in registers; B_t/C_t broadcast via LDS each step.
// ---------------------------------------------------------------------------
__global__ void k_scan(const float* __restrict__ xsT, const float* __restrict__ dt,
                       const float* __restrict__ xdbl, const float* __restrict__ A_log,
                       const float* __restrict__ Ds, float* __restrict__ ys,
                       int Bn, int L, int D, int R, int RN) {
  int kb = blockIdx.x;            // = k*Bn + b
  int k  = kb / Bn;
  int d  = threadIdx.x;
  __shared__ float sB[16], sC[16];
  float An[16], h[16];
  bool active = d < D;
  float Dval = 0.f;
  if (active) {
#pragma unroll
    for (int n = 0; n < 16; ++n) {
      An[n] = -__expf(A_log[((size_t)k * D + d) * 16 + n]);
      h[n]  = 0.f;
    }
    Dval = Ds[(size_t)k * D + d];
  }
  for (int t = 0; t < L; ++t) {
    size_t base = (size_t)kb * L + t;
    if (threadIdx.x < 32) {
      int n = threadIdx.x & 15;
      if (threadIdx.x < 16) sB[n] = xdbl[base * RN + R + n];
      else                  sC[n] = xdbl[base * RN + R + 16 + n];
    }
    __syncthreads();
    if (active) {
      float dtv = dt[base * D + d];
      float uv  = xsT[base * D + d];
      float du  = dtv * uv;
      float y = 0.f;
#pragma unroll
      for (int n = 0; n < 16; ++n) {
        h[n] = __expf(dtv * An[n]) * h[n] + du * sB[n];
        y += h[n] * sC[n];
      }
      ys[base * D + d] = y + Dval * uv;
    }
    __syncthreads();
  }
}

// ---------------------------------------------------------------------------
// Merge the 4 direction outputs back to token-major y[b*L+l][d].
// ---------------------------------------------------------------------------
__global__ void k_merge(const float* __restrict__ ys, float* __restrict__ y,
                        int Bn, int H, int W, int D) {
  int i = blockIdx.x * blockDim.x + threadIdx.x;
  int L = H * W;
  int total = Bn * L * D;
  if (i >= total) return;
  int d = i % D;
  int l = (i / D) % L;
  int b = i / (D * L);
  int h = l / W, w_ = l % W;
  int t1 = w_ * H + h;
  float v = ys[((size_t)(0 * Bn + b) * L + l)            * D + d]
          + ys[((size_t)(1 * Bn + b) * L + t1)           * D + d]
          + ys[((size_t)(2 * Bn + b) * L + (L - 1 - l))  * D + d]
          + ys[((size_t)(3 * Bn + b) * L + (L - 1 - t1)) * D + d];
  y[i] = v;
}

// ---------------------------------------------------------------------------
// Stride-2 3x3 conv + BN + ReLU, NHWC in -> NHWC out, optional NHWC add.
// ---------------------------------------------------------------------------
__global__ void k_conv_bn_relu(const float* __restrict__ x, const float* __restrict__ w,
                               const float* __restrict__ bc, const float* __restrict__ bg,
                               const float* __restrict__ bb, const float* __restrict__ bm,
                               const float* __restrict__ bv, const float* __restrict__ add,
                               float* __restrict__ out,
                               int Bn, int H, int W, int Cin, int Cout) {
  int Ho = H / 2, Wo = W / 2;
  int i = blockIdx.x * blockDim.x + threadIdx.x;
  int total = Bn * Ho * Wo * Cout;
  if (i >= total) return;
  int co = i % Cout;
  int wo = (i / Cout) % Wo;
  int ho = (i / (Cout * Wo)) % Ho;
  int b  = i / (Cout * Wo * Ho);
  float acc = bc[co];
  for (int ii = 0; ii < 3; ++ii) {
    int hi = 2 * ho + ii - 1;
    if (hi < 0 || hi >= H) continue;
    for (int jj = 0; jj < 3; ++jj) {
      int wi = 2 * wo + jj - 1;
      if (wi < 0 || wi >= W) continue;
      const float* xr = x + ((size_t)(b * H + hi) * W + wi) * Cin;
      const float* wr = w + ((size_t)co * Cin) * 9 + ii * 3 + jj;
      for (int ci = 0; ci < Cin; ++ci)
        acc += xr[ci] * wr[(size_t)ci * 9];
    }
  }
  float v = (acc - bm[co]) * rsqrtf(bv[co] + BN_EPS) * bg[co] + bb[co];
  v = fmaxf(v, 0.f);
  if (add) v += add[i];
  out[i] = v;
}

// ---------------------------------------------------------------------------
// NHWC -> NCHW (optional NCHW-layout add).
// ---------------------------------------------------------------------------
__global__ void k_nhwc_to_nchw(const float* __restrict__ in, const float* __restrict__ add,
                               float* __restrict__ out, int Bn, int C, int H, int W) {
  int i = blockIdx.x * blockDim.x + threadIdx.x;
  int total = Bn * C * H * W;
  if (i >= total) return;
  int w_ = i % W;
  int h  = (i / W) % H;
  int c  = (i / (W * H)) % C;
  int b  = i / (W * H * C);
  float v = in[((size_t)(b * H + h) * W + w_) * C + c];
  if (add) v += add[i];
  out[i] = v;
}

// ===========================================================================
// Host-side orchestration
// ===========================================================================
namespace {

struct VssP {
  const float *ln1_g, *ln1_b, *in_proj, *conv_w, *conv_b, *x_proj, *dt_w, *dt_b,
              *A_log, *Ds, *out_norm_g, *out_norm_b, *out_proj, *ln2_g, *ln2_b,
              *fc1_w, *fc1_b, *fc2_w, *fc2_b;
};
struct DcP { const float *w, *b, *bg, *bb, *bm, *bv; };

inline VssP load_vss(void* const* d_in, int base) {
  VssP p;
  p.ln1_g = (const float*)d_in[base + 0];  p.ln1_b = (const float*)d_in[base + 1];
  p.in_proj = (const float*)d_in[base + 2];
  p.conv_w = (const float*)d_in[base + 3]; p.conv_b = (const float*)d_in[base + 4];
  p.x_proj = (const float*)d_in[base + 5]; p.dt_w = (const float*)d_in[base + 6];
  p.dt_b = (const float*)d_in[base + 7];   p.A_log = (const float*)d_in[base + 8];
  p.Ds = (const float*)d_in[base + 9];
  p.out_norm_g = (const float*)d_in[base + 10];
  p.out_norm_b = (const float*)d_in[base + 11];
  p.out_proj = (const float*)d_in[base + 12];
  p.ln2_g = (const float*)d_in[base + 13]; p.ln2_b = (const float*)d_in[base + 14];
  p.fc1_w = (const float*)d_in[base + 15]; p.fc1_b = (const float*)d_in[base + 16];
  p.fc2_w = (const float*)d_in[base + 17]; p.fc2_b = (const float*)d_in[base + 18];
  return p;
}
inline DcP load_dc(void* const* d_in, int base) {
  DcP p;
  p.w = (const float*)d_in[base + 0];  p.b = (const float*)d_in[base + 1];
  p.bg = (const float*)d_in[base + 2]; p.bb = (const float*)d_in[base + 3];
  p.bm = (const float*)d_in[base + 4]; p.bv = (const float*)d_in[base + 5];
  return p;
}

inline void gemm(hipStream_t s, const float* A, const float* Wt, const float* bias,
                 const float* resid, float* out, int M, int N, int K, int lda, int act) {
  if ((M % 128 == 0) && (N % 32 == 0) && (K % 32 == 0) && (lda == K)) {
    dim3 grid(M / 128, N / 32);
    k_gemm_wmma_full<<<grid, 256, 0, s>>>(A, Wt, bias, resid, out, M, N, K, act);
  } else {
    dim3 grid((M + 63) / 64, (N + 31) / 32);
    k_gemm_wmma_edge<<<grid, 256, 0, s>>>(A, Wt, bias, resid, out, M, N, K, lda, act);
  }
}

inline int cdiv(int a, int b) { return (a + b - 1) / b; }

// One VSS block: x_out[T,C] = VSS(x_in[T,C]) with arena scratch.
void vss_block(hipStream_t s, const float* x_in, float* x_out, const VssP& p,
               int Bn, int H, int W, int C, float* arena) {
  const int T = Bn * H * W;
  const int D = 2 * C;
  const int L = H * W;
  const int R = (C + 15) / 16;
  const int RN = R + 32;

  size_t off = 0;
  auto alloc = [&](size_t n) { float* q = arena + off; off += n; return q; };
  float* ln1     = alloc((size_t)T * C);
  float* xz      = alloc((size_t)T * 2 * D);
  float* xconv   = alloc((size_t)T * D);
  float* xsT     = alloc((size_t)4 * T * D);
  float* xdbl    = alloc((size_t)4 * T * RN);
  float* dtbuf   = alloc((size_t)4 * T * D);
  float* ysbuf   = alloc((size_t)4 * T * D);
  float* ymerged = alloc((size_t)T * D);
  float* yact    = alloc((size_t)T * D);
  float* xafter  = alloc((size_t)T * C);
  float* ln2     = alloc((size_t)T * C);
  float* h1      = alloc((size_t)T * 4 * C);

  k_layernorm<<<cdiv(T, 8), 256, 0, s>>>(x_in, p.ln1_g, p.ln1_b, nullptr, 0, 0, ln1, T, C);
  gemm(s, ln1, p.in_proj, nullptr, nullptr, xz, T, 2 * D, C, C, 0);
  k_dwconv_silu<<<cdiv(T * D, 256), 256, 0, s>>>(xz, p.conv_w, p.conv_b, xconv, Bn, H, W, D);
  k_pack4<<<cdiv(T * D, 256), 256, 0, s>>>(xconv, xsT, Bn, H, W, D);
  for (int k = 0; k < 4; ++k)
    gemm(s, xsT + (size_t)k * T * D, p.x_proj + (size_t)k * RN * D, nullptr, nullptr,
         xdbl + (size_t)k * T * RN, T, RN, D, D, 0);
  for (int k = 0; k < 4; ++k)
    gemm(s, xdbl + (size_t)k * T * RN, p.dt_w + (size_t)k * D * R, p.dt_b + (size_t)k * D,
         nullptr, dtbuf + (size_t)k * T * D, T, D, R, RN, 2);
  k_scan<<<4 * Bn, D, 0, s>>>(xsT, dtbuf, xdbl, p.A_log, p.Ds, ysbuf, Bn, L, D, R, RN);
  k_merge<<<cdiv(T * D, 256), 256, 0, s>>>(ysbuf, ymerged, Bn, H, W, D);
  k_layernorm<<<cdiv(T, 8), 256, 0, s>>>(ymerged, p.out_norm_g, p.out_norm_b,
                                         xz, 2 * D, D, yact, T, D);
  gemm(s, yact, p.out_proj, nullptr, x_in, xafter, T, C, D, D, 0);
  k_layernorm<<<cdiv(T, 8), 256, 0, s>>>(xafter, p.ln2_g, p.ln2_b, nullptr, 0, 0, ln2, T, C);
  gemm(s, ln2, p.fc1_w, p.fc1_b, nullptr, h1, T, 4 * C, C, C, 1);
  gemm(s, h1, p.fc2_w, p.fc2_b, xafter, x_out, T, C, 4 * C, 4 * C, 0);
}

} // namespace

extern "C" void kernel_launch(void* const* d_in, const int* in_sizes, int n_in,
                              void* d_out, int out_size, void* d_ws, size_t ws_size,
                              hipStream_t stream) {
  (void)in_sizes; (void)n_in; (void)out_size; (void)ws_size;
  const float* b1   = (const float*)d_in[0];   // (4,32,32,128) NHWC
  const float* b2   = (const float*)d_in[1];   // (4,16,16,160) NHWC
  const float* b3   = (const float*)d_in[2];   // (4,8,8,320)   NHWC
  const float* bfin = (const float*)d_in[3];   // (4,256,4,4)   NCHW
  VssP v1 = load_vss(d_in, 4);
  DcP  c1 = load_dc(d_in, 23);
  VssP v2 = load_vss(d_in, 29);
  DcP  c2 = load_dc(d_in, 48);
  VssP v3 = load_vss(d_in, 54);
  DcP  c3 = load_dc(d_in, 73);

  float* out = (float*)d_out;
  float* t1_out = out;                           // (4,128,32,32) = 524288
  float* t2_out = out + 524288;                  // (4,160,16,16) = 163840
  float* t3_out = out + 524288 + 163840;         // (4,320,8,8)   = 81920
  float* t4_out = out + 524288 + 163840 + 81920; // (4,256,4,4)   = 16384

  float* ws = (float*)d_ws;
  size_t off = 0;
  auto alloc = [&](size_t n) { float* q = ws + off; off += n; return q; };
  float* s1_out = alloc((size_t)4 * 32 * 32 * 128);
  float* s2_in  = alloc((size_t)4 * 16 * 16 * 160);
  float* s2_out = alloc((size_t)4 * 16 * 16 * 160);
  float* s3_in  = alloc((size_t)4 * 8 * 8 * 320);
  float* s3_out = alloc((size_t)4 * 8 * 8 * 320);
  float* c3_out = alloc((size_t)4 * 4 * 4 * 256);
  float* arena  = ws + off;                      // reused across the 3 stages

  // -------- stage 1: C=128, 32x32 --------
  vss_block(stream, b1, s1_out, v1, 4, 32, 32, 128, arena);
  k_nhwc_to_nchw<<<cdiv(524288, 256), 256, 0, stream>>>(s1_out, nullptr, t1_out, 4, 128, 32, 32);
  k_conv_bn_relu<<<cdiv(4 * 16 * 16 * 160, 256), 256, 0, stream>>>(
      s1_out, c1.w, c1.b, c1.bg, c1.bb, c1.bm, c1.bv, b2, s2_in, 4, 32, 32, 128, 160);

  // -------- stage 2: C=160, 16x16 --------
  vss_block(stream, s2_in, s2_out, v2, 4, 16, 16, 160, arena);
  k_nhwc_to_nchw<<<cdiv(163840, 256), 256, 0, stream>>>(s2_out, nullptr, t2_out, 4, 160, 16, 16);
  k_conv_bn_relu<<<cdiv(4 * 8 * 8 * 320, 256), 256, 0, stream>>>(
      s2_out, c2.w, c2.b, c2.bg, c2.bb, c2.bm, c2.bv, b3, s3_in, 4, 16, 16, 160, 320);

  // -------- stage 3: C=320, 8x8 --------
  vss_block(stream, s3_in, s3_out, v3, 4, 8, 8, 320, arena);
  k_nhwc_to_nchw<<<cdiv(81920, 256), 256, 0, stream>>>(s3_out, nullptr, t3_out, 4, 320, 8, 8);
  k_conv_bn_relu<<<cdiv(4 * 4 * 4 * 256, 256), 256, 0, stream>>>(
      s3_out, c3.w, c3.b, c3.bg, c3.bb, c3.bm, c3.bv, nullptr, c3_out, 4, 8, 8, 320, 256);
  // t4 = conv3 (NCHW) + b
  k_nhwc_to_nchw<<<cdiv(16384, 256), 256, 0, stream>>>(c3_out, bfin, t4_out, 4, 256, 4, 4);
}
